// PositionAwareTopKPooling_12412455486097
// MI455X (gfx1250) — matrix operand order repaired and checked by reading, not compiled, not run on hardware
//
#include <hip/hip_runtime.h>
#include <hip/hip_bf16.h>

typedef _Float16 v16h __attribute__((ext_vector_type(16)));
typedef _Float16 v8h  __attribute__((ext_vector_type(8)));
typedef _Float16 v4h  __attribute__((ext_vector_type(4)));
typedef float    v8f  __attribute__((ext_vector_type(8)));

#define BATCH  4096
#define LSEQ   200
#define LPAD   208      // 13 tiles of 16
#define DDIM   64
#define TWOD   128
#define HID    128
#define OUTD   64
#define KSEL   50
#define KPAD   64       // 4 tiles of 16

// workspace byte offsets (all 256B aligned)
#define WS_W1I 0u
#define WS_W1E 32768u
#define WS_W2E 65536u
#define WS_POS 81920u   // 208*64 f16 = 26624 B -> total 108544 B

// LDS byte offsets
#define SM_A      0u        // 208*128 f16 = 53248
#define SM_SELA   53248u    // 64*128  f16 = 16384
#define SM_HE     69632u    // 64*128  f16 = 16384
#define SM_SCORE  86016u    // 208 f32     = 832
#define SM_FLAG   86848u    // 208 i32     = 832
#define SM_SEL    87680u    // 64 i32      = 256
#define SM_OACC   87936u    // 64 f32      = 256
#define SM_TOTAL  88192u

// ---- WMMA operand loaders (layouts per CDNA5 ISA 7.12.2, wave32) ----
// A (16x32 f16): lanes 0-15 row M=lane,   K = k0+[0..8) and k0+16+[0..8)
//                lanes16-31 row M=lane-16, K = k0+8+[0..8) and k0+24+[0..8)
__device__ __forceinline__ v16h load_a(const _Float16* base, int rowstride,
                                       int m0, int k0, int l16, int half) {
  const _Float16* p = base + (m0 + l16) * rowstride + k0 + half * 8;
  v8h lo = *(const v8h*)(p);
  v8h hi = *(const v8h*)(p + 16);
  return __builtin_shufflevector(lo, hi, 0,1,2,3,4,5,6,7,8,9,10,11,12,13,14,15);
}
// B (32x16 f16): lanes 0-15 col N=lane, K = k0+[0..8) and k0+16+[0..8); lanes 16-31 col N=lane-16, K = k0+8+[0..8) and k0+24+[0..8)
// Wt stored transposed: Wt[n][k], row stride 128 (K depth of every GEMM here)
__device__ __forceinline__ v16h load_b(const _Float16* wt, int n0, int k0, int l16, int half) {
  const _Float16* p = wt + (n0 + l16) * 128 + k0 + half * 8;
  v8h lo = *(const v8h*)(p);
  v8h hi = *(const v8h*)(p + 16);
  return __builtin_shufflevector(lo, hi, 0,1,2,3,4,5,6,7,8,9,10,11,12,13,14,15);
}

__device__ __forceinline__ v8f wmma_f16(v16h a, v16h b, v8f c) {
  return __builtin_amdgcn_wmma_f32_16x16x32_f16(false, a, false, b, (short)0, c, false, false);
}

__device__ __forceinline__ int rank_cmp(float v, float s, int j, int l) {
  return (int)(v > s) | ((int)(v == s) & (int)(j < l));
}

// ---------------- prep: f16-transposed weights + f16 pos table ----------------
__global__ void prep_kernel(const float* __restrict__ pos_table,
                            const float* __restrict__ w1i,
                            const float* __restrict__ w1e,
                            const float* __restrict__ w2e,
                            _Float16* __restrict__ W1i_t,
                            _Float16* __restrict__ W1e_t,
                            _Float16* __restrict__ W2e_t,
                            _Float16* __restrict__ Pos16) {
  int gid = blockIdx.x * blockDim.x + threadIdx.x;
  int stride = gridDim.x * blockDim.x;
  for (int i = gid; i < 128 * 128; i += stride) {
    int n = i >> 7, k = i & 127;
    W1i_t[n * 128 + k] = (_Float16)w1i[k * 128 + n];
    W1e_t[n * 128 + k] = (_Float16)w1e[k * 128 + n];
  }
  for (int i = gid; i < 64 * 128; i += stride) {
    int n = i >> 7, k = i & 127;
    W2e_t[n * 128 + k] = (_Float16)w2e[k * 64 + n];
  }
  for (int i = gid; i < LPAD * DDIM; i += stride) {
    int l = i >> 6;  // positions = min(arange(200), 9999) = arange(200)
    Pos16[i] = (l < LSEQ) ? (_Float16)pos_table[i] : (_Float16)0.0f;
  }
}

// ---------------- main: one workgroup (8 wave32) per batch row ----------------
__global__ __launch_bounds__(256) void topk_pool_kernel(
    const float* __restrict__ seq, const int* __restrict__ mask,
    const float* __restrict__ b1i, const float* __restrict__ w2i,
    const float* __restrict__ b2i, const float* __restrict__ b1e,
    const float* __restrict__ b2e,
    const _Float16* __restrict__ W1i_t, const _Float16* __restrict__ W1e_t,
    const _Float16* __restrict__ W2e_t, const _Float16* __restrict__ Pos16,
    float* __restrict__ out) {
  extern __shared__ char smem[];
  _Float16* A    = (_Float16*)(smem + SM_A);     // combined f16, 208x128
  _Float16* SELA = (_Float16*)(smem + SM_SELA);  // selected rows, 64x128
  _Float16* HE   = (_Float16*)(smem + SM_HE);    // relu hidden,   64x128
  float*    scr  = (float*)(smem + SM_SCORE);
  int*      flag = (int*)(smem + SM_FLAG);
  int*      sel  = (int*)(smem + SM_SEL);
  float*    oacc = (float*)(smem + SM_OACC);

  const int b    = blockIdx.x;
  const int tid  = threadIdx.x;
  const int lane = tid & 31;
  const int wave = tid >> 5;
  const int l16  = lane & 15;
  const int half = lane >> 4;

  // ---- stage combined = [seq | pos] in f16 ----
  const float* sp = seq + (size_t)b * LSEQ * DDIM;
  for (int i = tid; i < LSEQ * DDIM / 4; i += 256) {  // 3200 float4
    float4 v = ((const float4*)sp)[i];
    int l = i >> 4, c0 = (i & 15) * 4;
    v4h h; h.x = (_Float16)v.x; h.y = (_Float16)v.y; h.z = (_Float16)v.z; h.w = (_Float16)v.w;
    *(v4h*)(A + l * TWOD + c0) = h;
  }
  for (int i = tid; i < LPAD * DDIM / 4; i += 256) {  // pos half (rows>=200 are zero)
    v4h h = ((const v4h*)Pos16)[i];
    int l = i >> 4, c0 = (i & 15) * 4;
    *(v4h*)(A + l * TWOD + DDIM + c0) = h;
  }
  for (int i = tid; i < 8 * DDIM / 4; i += 256) {     // zero seq part of pad rows
    int l = LSEQ + (i >> 4), c0 = (i & 15) * 4;
    v4h z; z.x = (_Float16)0.f; z.y = (_Float16)0.f; z.z = (_Float16)0.f; z.w = (_Float16)0.f;
    *(v4h*)(A + l * TWOD + c0) = z;
  }
  __syncthreads();

  // ---- GEMM1: each wave owns whole M-tiles (all 8 N-tiles) -> deterministic,
  //      A operands hoisted into registers, one cross-lane reduce per M-tile ----
  const float b2i0 = b2i[0];
  for (int mt = wave; mt < 13; mt += 8) {
    const int m0 = mt * 16;
    const v16h a0 = load_a(A, TWOD, m0,  0, l16, half);
    const v16h a1 = load_a(A, TWOD, m0, 32, l16, half);
    const v16h a2 = load_a(A, TWOD, m0, 64, l16, half);
    const v16h a3 = load_a(A, TWOD, m0, 96, l16, half);
    v8f acc;
    #pragma unroll
    for (int j = 0; j < 8; ++j) acc[j] = 0.0f;
    #pragma unroll
    for (int nt = 0; nt < 8; ++nt) {
      const int n0 = nt * 16;
      const float bias = b1i[n0 + l16];
      v8f c;
      #pragma unroll
      for (int j = 0; j < 8; ++j) c[j] = bias;
      c = wmma_f16(a0, load_b(W1i_t, n0,  0, l16, half), c);
      c = wmma_f16(a1, load_b(W1i_t, n0, 32, l16, half), c);
      c = wmma_f16(a2, load_b(W1i_t, n0, 64, l16, half), c);
      c = wmma_f16(a3, load_b(W1i_t, n0, 96, l16, half), c);
      const float wv = w2i[n0 + l16];
      #pragma unroll
      for (int j = 0; j < 8; ++j) acc[j] += fmaxf(c[j], 0.0f) * wv;
    }
    #pragma unroll
    for (int j = 0; j < 8; ++j) {
      float r = acc[j];
      r += __shfl_xor(r, 1); r += __shfl_xor(r, 2);
      r += __shfl_xor(r, 4); r += __shfl_xor(r, 8);
      if (l16 == 0) scr[m0 + j + (half << 3)] = r + b2i0;  // plain store, no atomics
    }
  }
  __syncthreads();

  // ---- mask + pad sentinel ----
  const int* mrow = mask + (size_t)b * LSEQ;
  for (int l = tid; l < LPAD; l += 256) {
    float v = scr[l];
    if (l >= LSEQ) v = -3.0e38f;
    else if (mrow[l] == 0) v = -1.0e9f;
    scr[l] = v;
  }
  __syncthreads();

  // ---- top-50 via parallel stable ranking (matches lax.top_k tie-break) ----
  for (int l = tid; l < LPAD; l += 256) {
    const float s = scr[l];
    int r = 0;
    #pragma unroll 4
    for (int j4 = 0; j4 < LPAD; j4 += 4) {
      float4 v = *(const float4*)&scr[j4];
      r += rank_cmp(v.x, s, j4 + 0, l);
      r += rank_cmp(v.y, s, j4 + 1, l);
      r += rank_cmp(v.z, s, j4 + 2, l);
      r += rank_cmp(v.w, s, j4 + 3, l);
    }
    flag[l] = (r < KSEL) ? 1 : 0;
  }
  __syncthreads();
  // compact flagged indices in ascending order (== jnp.sort(idx))
  for (int l = tid; l < LPAD; l += 256) {
    if (flag[l]) {
      int pos = 0;
      const int l4 = l & ~3;
      for (int j = 0; j < l4; j += 4) {
        int4 f = *(const int4*)&flag[j];
        pos += f.x + f.y + f.z + f.w;
      }
      for (int j = l4; j < l; ++j) pos += flag[j];
      sel[pos] = l;
    }
  }
  __syncthreads();

  // ---- gather selected rows (pad 50->64 with zeros) ----
  {
    const unsigned* A32 = (const unsigned*)A;
    unsigned* S32 = (unsigned*)SELA;
    for (int i = tid; i < KPAD * 64; i += 256) {   // 64 dwords per 128-half row
      int r = i >> 6, c0 = i & 63;
      S32[i] = (r < KSEL) ? A32[sel[r] * 64 + c0] : 0u;
    }
  }
  __syncthreads();

  // ---- GEMM2a: HE = relu(SELA @ W1e + b1e) ----
  for (int t = wave; t < 4 * 8; t += 8) {
    const int m0 = (t >> 3) * 16, n0 = (t & 7) * 16;
    const float bias = b1e[n0 + l16];
    v8f c;
    #pragma unroll
    for (int j = 0; j < 8; ++j) c[j] = bias;
    #pragma unroll
    for (int k0 = 0; k0 < 128; k0 += 32)
      c = wmma_f16(load_a(SELA, TWOD, m0, k0, l16, half), load_b(W1e_t, n0, k0, l16, half), c);
    #pragma unroll
    for (int j = 0; j < 8; ++j) {
      int row = m0 + j + (half << 3);
      HE[row * HID + n0 + l16] = (_Float16)fmaxf(c[j], 0.0f);
    }
  }
  __syncthreads();

  // ---- GEMM2b: enc = HE @ W2e + b2e, mean over valid 50 rows.
  //      wave w<4 owns N-column w across all M-tiles -> register accumulate, no atomics ----
  if (wave < 4) {
    const int n0 = wave * 16;
    const float bias = b2e[n0 + l16];
    float ptot = 0.0f;
    #pragma unroll
    for (int mt = 0; mt < 4; ++mt) {
      const int m0 = mt * 16;
      v8f c;
      #pragma unroll
      for (int j = 0; j < 8; ++j) c[j] = bias;
      #pragma unroll
      for (int k0 = 0; k0 < 128; k0 += 32)
        c = wmma_f16(load_a(HE, HID, m0, k0, l16, half), load_b(W2e_t, n0, k0, l16, half), c);
      #pragma unroll
      for (int j = 0; j < 8; ++j) {
        int row = m0 + j + (half << 3);
        if (row < KSEL) ptot += c[j];
      }
    }
    ptot += __shfl_xor(ptot, 16);              // combine the two row-halves of column n
    if (half == 0) oacc[n0 + l16] = ptot;
  }
  __syncthreads();

  if (tid < OUTD) out[(size_t)b * OUTD + tid] = oacc[tid] * (1.0f / (float)KSEL);
}

extern "C" void kernel_launch(void* const* d_in, const int* in_sizes, int n_in,
                              void* d_out, int out_size, void* d_ws, size_t ws_size,
                              hipStream_t stream) {
  const float* seq = (const float*)d_in[0];
  const int*   msk = (const int*)d_in[1];
  const float* pos = (const float*)d_in[2];
  const float* w1i = (const float*)d_in[3];
  const float* b1i = (const float*)d_in[4];
  const float* w2i = (const float*)d_in[5];
  const float* b2i = (const float*)d_in[6];
  const float* w1e = (const float*)d_in[7];
  const float* b1e = (const float*)d_in[8];
  const float* w2e = (const float*)d_in[9];
  const float* b2e = (const float*)d_in[10];
  float* out = (float*)d_out;

  char* ws = (char*)d_ws;
  _Float16* W1i_t = (_Float16*)(ws + WS_W1I);
  _Float16* W1e_t = (_Float16*)(ws + WS_W1E);
  _Float16* W2e_t = (_Float16*)(ws + WS_W2E);
  _Float16* Pos16 = (_Float16*)(ws + WS_POS);

  prep_kernel<<<64, 256, 0, stream>>>(pos, w1i, w1e, w2e, W1i_t, W1e_t, W2e_t, Pos16);
  topk_pool_kernel<<<BATCH, 256, SM_TOTAL, stream>>>(
      seq, msk, b1i, w2i, b2i, b1e, b2e, W1i_t, W1e_t, W2e_t, Pos16, out);
}
